// MultiHeadAttention_9148280340574
// MI455X (gfx1250) — matrix operand order repaired and checked by reading, not compiled, not run on hardware
//
#include <hip/hip_runtime.h>
#include <stdint.h>

// ---------------------------------------------------------------------------
// MHA for MI455X (gfx1250, wave32).
//  * All matmuls through v_wmma_f32_16x16x32_bf16 (bf16 A/B, f32 accum).
//  * Global->LDS staging of bf16 tiles via the Tensor Data Mover
//    (tensor_load_to_lds + s_wait_tensorcnt), double-buffered in the GEMMs.
//  * f32 activations are converted to bf16 while staging (TDM cannot convert),
//    using a register double-buffer so the global loads issue a K-step ahead.
//
// Roofline: ~137 GFLOP vs ~200 MB HBM traffic @ 23.3 TB/s -> matrix pipes
// mandatory; bf16+f32-accum is the right precision at this intensity.
// ---------------------------------------------------------------------------

typedef __bf16 v16bf __attribute__((ext_vector_type(16)));
typedef __bf16 v8bf  __attribute__((ext_vector_type(8)));
typedef __bf16 v4bf  __attribute__((ext_vector_type(4)));
typedef __bf16 v2bf  __attribute__((ext_vector_type(2)));
typedef float  v8f   __attribute__((ext_vector_type(8)));
typedef unsigned int u32x4 __attribute__((ext_vector_type(4)));
typedef int          i32x4 __attribute__((ext_vector_type(4)));
typedef int          i32x8 __attribute__((ext_vector_type(8)));

#define D_MODEL  1024
#define N_HEADS  16
#define HEAD_DIM 64
#define SEQ      2048
#define BATCH    4

#if defined(__has_builtin)
#if __has_builtin(__builtin_amdgcn_tensor_load_to_lds)
#define HAVE_TDM 1
#endif
#endif

static __device__ inline v16bf make_frag(v8bf lo, v8bf hi) {
  v16bf r;
#pragma unroll
  for (int i = 0; i < 8; ++i) { r[i] = lo[i]; r[8 + i] = hi[i]; }
  return r;
}

// Issue a TDM load of a 2-D bf16 tile (tile_d0 x tile_d1 elements, row stride
// stride0 elements) from global into LDS. Call from ONE wave; follow with
// tdm_wait() + workgroup barrier before consumers read the tile.
static __device__ inline void tdm_load_2d(__bf16* lds_ptr, const __bf16* gptr,
                                          unsigned tile_d0, unsigned tile_d1,
                                          unsigned tensor_d0, unsigned tensor_d1,
                                          unsigned long long stride0) {
#ifdef HAVE_TDM
  unsigned long long ga = (unsigned long long)(uintptr_t)gptr;
  unsigned lds_addr = (unsigned)(uintptr_t)lds_ptr;  // flat->LDS = addr[31:0]
  u32x4 g0;
  g0[0] = 1u;                                          // count=1 (valid, user)
  g0[1] = lds_addr;                                    // lds_addr
  g0[2] = (unsigned)(ga & 0xFFFFFFFFu);                // global_addr[31:0]
  g0[3] = (unsigned)((ga >> 32) & 0x01FFFFFFu) | (2u << 30);  // [56:32]|type=2
  i32x8 g1;
  g1[0] = (int)(1u << 16);                             // wg_mask=0, data_size=2B
  g1[1] = (int)((tensor_d0 & 0xFFFFu) << 16);          // tensor_dim0[15:0]
  g1[2] = (int)(((tensor_d0 >> 16) & 0xFFFFu) | ((tensor_d1 & 0xFFFFu) << 16));
  g1[3] = (int)(((tensor_d1 >> 16) & 0xFFFFu) | ((tile_d0 & 0xFFFFu) << 16));
  g1[4] = (int)(tile_d1 & 0xFFFFu);                    // tile_dim1, tile_dim2=0
  g1[5] = (int)(unsigned)(stride0 & 0xFFFFFFFFu);      // dim0 stride lo
  g1[6] = (int)(unsigned)((stride0 >> 32) & 0xFFFFu);  // stride hi, stride1=0
  g1[7] = 0;
  i32x4 z4 = {0, 0, 0, 0};
#if __clang_major__ >= 23
  i32x8 z8 = {0, 0, 0, 0, 0, 0, 0, 0};
  __builtin_amdgcn_tensor_load_to_lds(g0, g1, z4, z4, z8, 0);
#else
  __builtin_amdgcn_tensor_load_to_lds(g0, g1, z4, z4, 0);
#endif
#else
  // Fallback: wave-cooperative synchronous copy.
  int lane = threadIdx.x & 31;
  for (unsigned r = 0; r < tile_d1; ++r)
    for (unsigned c = (unsigned)lane * 8; c < tile_d0; c += 32 * 8)
      *(v8bf*)(lds_ptr + (size_t)r * tile_d0 + c) =
          *(const v8bf*)(gptr + (size_t)r * stride0 + c);
#endif
}

static __device__ inline void tdm_wait() {
#ifdef HAVE_TDM
  __builtin_amdgcn_s_wait_tensorcnt(0);
#endif
}

// ------------------------- weight f32 -> bf16 ------------------------------
__global__ __launch_bounds__(256) void cvt_f32_bf16(
    const float* __restrict__ s, __bf16* __restrict__ d, int n) {
  int i = (blockIdx.x * 256 + threadIdx.x) * 4;
  if (i + 3 < n) {
    float4 v = *(const float4*)(s + i);
    v4bf o;
    o[0] = (__bf16)v.x; o[1] = (__bf16)v.y;
    o[2] = (__bf16)v.z; o[3] = (__bf16)v.w;
    *(v4bf*)(d + i) = o;  // 8B store
  }
}

// ------------------------- projection GEMM ---------------------------------
// C[m,n] = sum_k A[m,k] * W[n,k] + bias[n]    (x @ W^T + b)
// Block tile 128x128, K-step 32, 8 waves each computing 32x64.
// Double-buffered LDS; W tiles (and A tiles when bf16) staged by TDM.
template <bool A_F32, bool SPLIT_HEADS>
__global__ __launch_bounds__(256) void proj_gemm(
    const void* __restrict__ Aptr, const __bf16* __restrict__ Wbf,
    const float* __restrict__ bias, void* __restrict__ Cptr) {
  __shared__ __bf16 As[2][128][32];
  __shared__ __bf16 Bs[2][128][32];

  const int t    = threadIdx.x;
  const int lane = t & 31;
  const int w    = t >> 5;
  const int hl   = lane >> 4;   // 16-lane half
  const int ln   = lane & 15;
  const int m0   = blockIdx.y * 128;
  const int n0   = blockIdx.x * 128;
  const int wm   = (w >> 1) * 32;   // wave M offset (4 rows of waves)
  const int wn   = (w & 1) * 64;    // wave N offset (2 cols of waves)

  v8f acc[2][4];
#pragma unroll
  for (int i = 0; i < 2; ++i)
#pragma unroll
    for (int j = 0; j < 4; ++j) acc[i][j] = {};

  const int ldRow = t >> 1;
  const int ldSeg = (t & 1) * 16;
  const float* Af32 = (const float*)Aptr;
  const __bf16* Abf = (const __bf16*)Aptr;

  float4 xr[4];  // A register double-buffer (f32 path)

  // ---- prologue: stage tile 0 ----
  if (w == 0)
    tdm_load_2d(&Bs[0][0][0], Wbf + (size_t)n0 * D_MODEL, 32, 128,
                D_MODEL, D_MODEL, D_MODEL);
  if (A_F32) {
    const float4* p = (const float4*)(Af32 + (size_t)(m0 + ldRow) * D_MODEL + ldSeg);
    xr[0] = p[0]; xr[1] = p[1]; xr[2] = p[2]; xr[3] = p[3];
    v8bf lo, hi;
    lo[0] = (__bf16)xr[0].x; lo[1] = (__bf16)xr[0].y; lo[2] = (__bf16)xr[0].z;
    lo[3] = (__bf16)xr[0].w; lo[4] = (__bf16)xr[1].x; lo[5] = (__bf16)xr[1].y;
    lo[6] = (__bf16)xr[1].z; lo[7] = (__bf16)xr[1].w;
    hi[0] = (__bf16)xr[2].x; hi[1] = (__bf16)xr[2].y; hi[2] = (__bf16)xr[2].z;
    hi[3] = (__bf16)xr[2].w; hi[4] = (__bf16)xr[3].x; hi[5] = (__bf16)xr[3].y;
    hi[6] = (__bf16)xr[3].z; hi[7] = (__bf16)xr[3].w;
    *(v8bf*)&As[0][ldRow][ldSeg]     = lo;
    *(v8bf*)&As[0][ldRow][ldSeg + 8] = hi;
  } else if (w == 1) {
    tdm_load_2d(&As[0][0][0], Abf + (size_t)m0 * D_MODEL, 32, 128,
                D_MODEL, D_MODEL, D_MODEL);
  }
  if (w == 0 || (!A_F32 && w == 1)) tdm_wait();
  __syncthreads();

  const int NIT = D_MODEL / 32;
  for (int it = 0; it < NIT; ++it) {
    const int buf = it & 1;
    const bool has_next = (it + 1) < NIT;
    const int knext = (it + 1) * 32;

    // ---- kick off next tile (async) ----
    if (has_next) {
      if (w == 0)
        tdm_load_2d(&Bs[buf ^ 1][0][0],
                    Wbf + (size_t)n0 * D_MODEL + knext, 32, 128,
                    D_MODEL, D_MODEL, D_MODEL);
      if (A_F32) {
        const float4* p =
            (const float4*)(Af32 + (size_t)(m0 + ldRow) * D_MODEL + knext + ldSeg);
        xr[0] = p[0]; xr[1] = p[1]; xr[2] = p[2]; xr[3] = p[3];
      } else if (w == 1) {
        tdm_load_2d(&As[buf ^ 1][0][0],
                    Abf + (size_t)m0 * D_MODEL + knext, 32, 128,
                    D_MODEL, D_MODEL, D_MODEL);
      }
    }

    // ---- fragments + WMMA on current buffer ----
    const int akb = hl * 8;   // A frag: lanes<16 take k 0-7/16-23, else 8-15/24-31
    v16bf af[2];
#pragma unroll
    for (int i = 0; i < 2; ++i) {
      int r = wm + i * 16 + ln;
      af[i] = make_frag(*(const v8bf*)&As[buf][r][akb],
                        *(const v8bf*)&As[buf][r][akb + 16]);
    }
    const int bkb = hl * 8;   // B frag: same K order as A frag
#pragma unroll
    for (int j = 0; j < 4; ++j) {
      int n = wn + j * 16 + ln;
      v16bf bf_ = make_frag(*(const v8bf*)&Bs[buf][n][bkb],
                            *(const v8bf*)&Bs[buf][n][bkb + 16]);
#pragma unroll
      for (int i = 0; i < 2; ++i)
        acc[i][j] = __builtin_amdgcn_wmma_f32_16x16x32_bf16(
            false, af[i], false, bf_, (short)0, acc[i][j], false, false);
    }

    // ---- publish next buffer ----
    if (has_next) {
      if (A_F32) {
        v8bf lo, hi;
        lo[0] = (__bf16)xr[0].x; lo[1] = (__bf16)xr[0].y; lo[2] = (__bf16)xr[0].z;
        lo[3] = (__bf16)xr[0].w; lo[4] = (__bf16)xr[1].x; lo[5] = (__bf16)xr[1].y;
        lo[6] = (__bf16)xr[1].z; lo[7] = (__bf16)xr[1].w;
        hi[0] = (__bf16)xr[2].x; hi[1] = (__bf16)xr[2].y; hi[2] = (__bf16)xr[2].z;
        hi[3] = (__bf16)xr[2].w; hi[4] = (__bf16)xr[3].x; hi[5] = (__bf16)xr[3].y;
        hi[6] = (__bf16)xr[3].z; hi[7] = (__bf16)xr[3].w;
        *(v8bf*)&As[buf ^ 1][ldRow][ldSeg]     = lo;
        *(v8bf*)&As[buf ^ 1][ldRow][ldSeg + 8] = hi;
      }
      if (w == 0 || (!A_F32 && w == 1)) tdm_wait();
      __syncthreads();
    }
  }

  // ---- epilogue: bias + store ----
#pragma unroll
  for (int j = 0; j < 4; ++j) {
    int n = n0 + wn + j * 16 + ln;
    float bv = bias[n];
#pragma unroll
    for (int i = 0; i < 2; ++i) {
#pragma unroll
      for (int e = 0; e < 8; ++e) {   // C layout: elem e -> row e + hl*8
        int m = m0 + wm + i * 16 + hl * 8 + e;
        float val = acc[i][j][e] + bv;
        if (SPLIT_HEADS) {
          int b = m >> 11, s = m & (SEQ - 1);
          int h = n >> 6, d = n & (HEAD_DIM - 1);
          ((__bf16*)Cptr)[((size_t)(b * N_HEADS + h) * SEQ + s) * HEAD_DIM + d] =
              (__bf16)val;
        } else {
          ((float*)Cptr)[(size_t)m * D_MODEL + n] = val;
        }
      }
    }
  }
}

// ------------------------- flash attention ---------------------------------
// grid.y = b*H+h, grid.x = q-block (128 rows). 8 waves x 16 q-rows each,
// streaming 128-key chunks with online softmax, all matmuls via WMMA.
// Q block and K chunks are contiguous 16KB blocks -> staged by TDM.
__global__ __launch_bounds__(256) void attention_kernel(
    const __bf16* __restrict__ Qh, const __bf16* __restrict__ Kh,
    const __bf16* __restrict__ Vh, __bf16* __restrict__ Oc) {
  __shared__ __bf16 Qs[128][HEAD_DIM];   // 16 KB
  __shared__ __bf16 Ks[128][HEAD_DIM];   // 16 KB
  __shared__ __bf16 VTs[HEAD_DIM][128];  // 16 KB, V transposed
  __shared__ __bf16 Ps[8][16][128];      // 32 KB, per-wave P tiles

  const int t    = threadIdx.x;
  const int lane = t & 31;
  const int w    = t >> 5;
  const int hl   = lane >> 4;
  const int ln   = lane & 15;
  const int bh   = blockIdx.y;
  const int q0   = blockIdx.x * 128;

  const __bf16* Qb = Qh + ((size_t)bh * SEQ + q0) * HEAD_DIM;
  const __bf16* Kb = Kh + (size_t)bh * SEQ * HEAD_DIM;
  const __bf16* Vb = Vh + (size_t)bh * SEQ * HEAD_DIM;

  // Q block is one contiguous 128*64-element run: single TDM copy.
  if (w == 0)
    tdm_load_2d(&Qs[0][0], Qb, 128 * HEAD_DIM, 1, 128 * HEAD_DIM, 1,
                128 * HEAD_DIM);

  float mrow[8], lrow[8];
  v8f accO[4];
#pragma unroll
  for (int j = 0; j < 8; ++j) { mrow[j] = -1e30f; lrow[j] = 0.f; }
#pragma unroll
  for (int dt = 0; dt < 4; ++dt) accO[dt] = {};

  for (int kc = 0; kc < SEQ; kc += 128) {
    __syncthreads();  // previous chunk fully consumed before restage
    // ---- K chunk: contiguous 16KB -> TDM (overlaps the V staging below)
    if (w == 0)
      tdm_load_2d(&Ks[0][0], Kb + (size_t)kc * HEAD_DIM, 128 * HEAD_DIM, 1,
                  128 * HEAD_DIM, 1, 128 * HEAD_DIM);
    // ---- stage V chunk transposed: VTs[d][k] = V[kc+k][d]
    {
      int kv0  = (t >> 2) * 2;
      int dseg = (t & 3) * 16;
      const __bf16* r0 = Vb + (size_t)(kc + kv0) * HEAD_DIM + dseg;
      const __bf16* r1 = r0 + HEAD_DIM;
      v8bf a0 = ((const v8bf*)r0)[0], a1 = ((const v8bf*)r0)[1];
      v8bf b0 = ((const v8bf*)r1)[0], b1 = ((const v8bf*)r1)[1];
#pragma unroll
      for (int e = 0; e < 8; ++e) {
        v2bf x; x[0] = a0[e]; x[1] = b0[e];
        *(v2bf*)&VTs[dseg + e][kv0] = x;
        v2bf y; y[0] = a1[e]; y[1] = b1[e];
        *(v2bf*)&VTs[dseg + 8 + e][kv0] = y;
      }
    }
    if (w == 0) tdm_wait();  // Q (first iter) + K chunk complete
    __syncthreads();

    // ---- scores: S = Q K^T (16 q-rows x 128 keys per wave) ----
    v16bf aq[2];
#pragma unroll
    for (int ks = 0; ks < 2; ++ks) {
      int r  = w * 16 + ln;
      int kb = ks * 32 + hl * 8;
      aq[ks] = make_frag(*(const v8bf*)&Qs[r][kb],
                         *(const v8bf*)&Qs[r][kb + 16]);
    }
    v8f st[8];
#pragma unroll
    for (int nt = 0; nt < 8; ++nt) {
      v8f s = {};
#pragma unroll
      for (int ks = 0; ks < 2; ++ks) {
        int n  = nt * 16 + ln;
        int kb = ks * 32 + hl * 8;
        v16bf bk = make_frag(*(const v8bf*)&Ks[n][kb],
                             *(const v8bf*)&Ks[n][kb + 16]);
        s = __builtin_amdgcn_wmma_f32_16x16x32_bf16(
            false, aq[ks], false, bk, (short)0, s, false, false);
      }
      st[nt] = s;
    }

    // ---- online softmax (row j lives at C-elem j, rows split by lane half)
    const float scale = 0.125f;  // 1/sqrt(64)
#pragma unroll
    for (int j = 0; j < 8; ++j) {
      float rmax = -1e30f;
#pragma unroll
      for (int nt = 0; nt < 8; ++nt) {
        st[nt][j] *= scale;
        rmax = fmaxf(rmax, st[nt][j]);
      }
      rmax = fmaxf(rmax, __shfl_xor(rmax, 1, 32));
      rmax = fmaxf(rmax, __shfl_xor(rmax, 2, 32));
      rmax = fmaxf(rmax, __shfl_xor(rmax, 4, 32));
      rmax = fmaxf(rmax, __shfl_xor(rmax, 8, 32));
      float mnew = fmaxf(mrow[j], rmax);
      float corr = __expf(mrow[j] - mnew);
      mrow[j] = mnew;
      float psum = 0.f;
#pragma unroll
      for (int nt = 0; nt < 8; ++nt) {
        float p = __expf(st[nt][j] - mnew);
        psum += p;
        Ps[w][hl * 8 + j][nt * 16 + ln] = (__bf16)p;
      }
      psum += __shfl_xor(psum, 1, 32);
      psum += __shfl_xor(psum, 2, 32);
      psum += __shfl_xor(psum, 4, 32);
      psum += __shfl_xor(psum, 8, 32);
      lrow[j] = lrow[j] * corr + psum;
#pragma unroll
      for (int dt = 0; dt < 4; ++dt) accO[dt][j] *= corr;
    }

    // ---- O += P V (per-wave LDS round-trip re-lays P into A fragments) ----
#pragma unroll
    for (int ks = 0; ks < 4; ++ks) {
      int kb = ks * 32 + hl * 8;
      v16bf ap = make_frag(*(const v8bf*)&Ps[w][ln][kb],
                           *(const v8bf*)&Ps[w][ln][kb + 16]);
#pragma unroll
      for (int dt = 0; dt < 4; ++dt) {
        int n  = dt * 16 + ln;
        int vb = ks * 32 + hl * 8;
        v16bf bv = make_frag(*(const v8bf*)&VTs[n][vb],
                             *(const v8bf*)&VTs[n][vb + 16]);
        accO[dt] = __builtin_amdgcn_wmma_f32_16x16x32_bf16(
            false, ap, false, bv, (short)0, accO[dt], false, false);
      }
    }
  }

  // ---- normalize + store to [B,S,D] bf16 ----
  const int b = bh >> 4, h = bh & 15;
#pragma unroll
  for (int dt = 0; dt < 4; ++dt) {
#pragma unroll
    for (int j = 0; j < 8; ++j) {
      int r = q0 + w * 16 + hl * 8 + j;
      int c = h * HEAD_DIM + dt * 16 + ln;
      float val = accO[dt][j] / lrow[j];
      Oc[((size_t)b * SEQ + r) * D_MODEL + c] = (__bf16)val;
    }
  }
}

// ------------------------------- launch ------------------------------------
extern "C" void kernel_launch(void* const* d_in, const int* in_sizes, int n_in,
                              void* d_out, int out_size, void* d_ws,
                              size_t ws_size, hipStream_t stream) {
  (void)in_sizes; (void)n_in; (void)out_size; (void)ws_size;

  const float* query = (const float*)d_in[0];
  const float* key   = (const float*)d_in[1];
  const float* value = (const float*)d_in[2];
  const float* Wq = (const float*)d_in[3];
  const float* bq = (const float*)d_in[4];
  const float* Wk = (const float*)d_in[5];
  const float* bk = (const float*)d_in[6];
  const float* Wv = (const float*)d_in[7];
  const float* bv = (const float*)d_in[8];
  const float* Wo = (const float*)d_in[9];
  const float* bo = (const float*)d_in[10];
  float* out = (float*)d_out;

  char* ws = (char*)d_ws;
  const size_t W_BYTES = (size_t)D_MODEL * D_MODEL * 2;                 // 2 MB
  const size_t H_BYTES = (size_t)BATCH * N_HEADS * SEQ * HEAD_DIM * 2;  // 16 MB
  __bf16* wq_bf = (__bf16*)(ws);
  __bf16* wk_bf = (__bf16*)(ws + W_BYTES);
  __bf16* wv_bf = (__bf16*)(ws + 2 * W_BYTES);
  __bf16* wo_bf = (__bf16*)(ws + 3 * W_BYTES);
  __bf16* Qh    = (__bf16*)(ws + 4 * W_BYTES);
  __bf16* Kh    = (__bf16*)(ws + 4 * W_BYTES + H_BYTES);
  __bf16* Vh    = (__bf16*)(ws + 4 * W_BYTES + 2 * H_BYTES);
  __bf16* Ocws  = (__bf16*)(ws + 4 * W_BYTES + 3 * H_BYTES);

  const int WN = D_MODEL * D_MODEL;
  cvt_f32_bf16<<<WN / 1024, 256, 0, stream>>>(Wq, wq_bf, WN);
  cvt_f32_bf16<<<WN / 1024, 256, 0, stream>>>(Wk, wk_bf, WN);
  cvt_f32_bf16<<<WN / 1024, 256, 0, stream>>>(Wv, wv_bf, WN);
  cvt_f32_bf16<<<WN / 1024, 256, 0, stream>>>(Wo, wo_bf, WN);

  dim3 ggrid(D_MODEL / 128, (BATCH * SEQ) / 128);  // (8, 64)
  proj_gemm<true, true><<<ggrid, 256, 0, stream>>>(query, wq_bf, bq, Qh);
  proj_gemm<true, true><<<ggrid, 256, 0, stream>>>(key,   wk_bf, bk, Kh);
  proj_gemm<true, true><<<ggrid, 256, 0, stream>>>(value, wv_bf, bv, Vh);

  attention_kernel<<<dim3(SEQ / 128, BATCH * N_HEADS), 256, 0, stream>>>(
      Qh, Kh, Vh, Ocws);

  proj_gemm<false, false><<<ggrid, 256, 0, stream>>>(Ocws, wo_bf, bo, out);
}
